// ResBlock_71021579207010
// MI455X (gfx1250) — 8-bit matrix operand order repaired and checked by reading, compiled, not run on hardware
//
#include <hip/hip_runtime.h>

typedef __attribute__((ext_vector_type(8))) int v8i;

#define T_DIM   8192
#define C_DIM   128
#define B_DIM   32
#define TILE_T  128
#define NTHREADS 256


__global__ __launch_bounds__(256) void prep_sign_wsc(const float* __restrict__ wsc,
                                                     signed char* __restrict__ wscb) {
    int g = blockIdx.x * 256 + threadIdx.x;
    if (g < C_DIM * C_DIM)
        wscb[g] = (wsc[g] >= 0.0f) ? (signed char)1 : (signed char)-1;
}

__global__ __launch_bounds__(128) void prep_ck(const float* __restrict__ w2,
                                               const float* __restrict__ beta2,
                                               const float* __restrict__ mean2,
                                               const float* __restrict__ var2,
                                               float* __restrict__ ck) {
    int f = threadIdx.x;
    float s0 = 0.0f, s1 = 0.0f;
    for (int c = 0; c < C_DIM; ++c) {
        s0 += (w2[(0 * C_DIM + c) * C_DIM + f] >= 0.0f) ? 1.0f : -1.0f;
        s1 += (w2[(1 * C_DIM + c) * C_DIM + f] >= 0.0f) ? 1.0f : -1.0f;
    }
    float inv = rsqrtf(var2[f] + 1e-3f);
    float hA = (s1      - mean2[f]) * inv + beta2[f];
    float hB = (s0 + s1 - mean2[f]) * inv + beta2[f];
    ck[f]          = fmaxf(hA, 0.0f);
    ck[C_DIM + f]  = fmaxf(hB, 0.0f);
}

__global__ __launch_bounds__(NTHREADS) void resblock_shortcut_wmma(
        const float* __restrict__ x,
        const signed char* __restrict__ wscb,
        const float* __restrict__ ck,
        float* __restrict__ out) {

    __shared__ __align__(16) signed char Xs[TILE_T * C_DIM];

    const int tid  = threadIdx.x;
    const int lane = tid & 31;
    const int wave = tid >> 5;
    const long long rowBase = (long long)blockIdx.x * TILE_T;
    const int t0 = (int)(rowBase & (long long)(T_DIM - 1));

    const float4* xin = (const float4*)(x + rowBase * C_DIM);
    for (int i = tid; i < TILE_T * (C_DIM / 4); i += NTHREADS) {
        float4 v = xin[i];
        unsigned int b0 = (v.x >= 0.0f) ? 0x01u : 0xFFu;
        unsigned int b1 = (v.y >= 0.0f) ? 0x01u : 0xFFu;
        unsigned int b2 = (v.z >= 0.0f) ? 0x01u : 0xFFu;
        unsigned int b3 = (v.w >= 0.0f) ? 0x01u : 0xFFu;
        ((unsigned int*)Xs)[i] = b0 | (b1 << 8) | (b2 << 16) | (b3 << 24);
    }
    __syncthreads();

    const int n0 = wave * 16;
    v8i bw[2];
    #pragma unroll
    for (int kc = 0; kc < 2; ++kc) {
        const unsigned char* cb = (const unsigned char*)wscb + (kc * 64 + (lane >> 4) * 8) * C_DIM + n0 + (lane & 15);
        v8i b;
        #pragma unroll
        for (int j = 0; j < 8; ++j) {
            const int kk = (j >> 1) * 16 + (j & 1) * 4;
            b[j] = (int)((unsigned int)cb[kk * C_DIM] | ((unsigned int)cb[(kk + 1) * C_DIM] << 8) |
                         ((unsigned int)cb[(kk + 2) * C_DIM] << 16) | ((unsigned int)cb[(kk + 3) * C_DIM] << 24));
        }
        bw[kc] = b;
    }

    const int nlane = n0 + (lane & 15);
    const float ckLo = ck[nlane];
    const float ckHi = ck[C_DIM + nlane];
    const int m  = lane & 15;
    const int kb = (lane >> 4) * 8;
    const int mhi = (lane >> 4) * 8;

    float* outp = out + rowBase * C_DIM;

    #pragma unroll 2
    for (int mt = 0; mt < TILE_T / 16; ++mt) {
        v8i acc = {0, 0, 0, 0, 0, 0, 0, 0};
        #pragma unroll
        for (int kc = 0; kc < 2; ++kc) {
            const signed char* row = &Xs[(mt * 16 + m) * C_DIM + kc * 64];
            v8i a;
            const int* q0 = (const int*)(row + kb);
            const int* q1 = (const int*)(row + kb + 16);
            const int* q2 = (const int*)(row + kb + 32);
            const int* q3 = (const int*)(row + kb + 48);
            a[0] = q0[0]; a[1] = q0[1];
            a[2] = q1[0]; a[3] = q1[1];
            a[4] = q2[0]; a[5] = q2[1];
            a[6] = q3[0]; a[7] = q3[1];
            acc = __builtin_amdgcn_wmma_i32_16x16x64_iu8(true, a, true, bw[kc], acc,
                                                         false, false);
        }
        #pragma unroll
        for (int r = 0; r < 8; ++r) {
            int mrow = mt * 16 + mhi + r;
            int t    = t0 + mrow;
            float v  = (float)acc[r] + ((t >= 16) ? ckHi : ckLo);
            outp[(long long)mrow * C_DIM + nlane] = (v > 0.0f) ? v : 0.0f;
        }
    }
}

extern "C" void kernel_launch(void* const* d_in, const int* in_sizes, int n_in,
                              void* d_out, int out_size, void* d_ws, size_t ws_size,
                              hipStream_t stream) {
    const float* x     = (const float*)d_in[0];
    const float* w2    = (const float*)d_in[2];
    const float* wsc   = (const float*)d_in[3];
    const float* beta2 = (const float*)d_in[7];
    const float* mean2 = (const float*)d_in[8];
    const float* var2  = (const float*)d_in[9];

    signed char* wscb = (signed char*)d_ws;
    float*       ckp  = (float*)((char*)d_ws + 16384);

    prep_sign_wsc<<<64, 256, 0, stream>>>(wsc, wscb);
    prep_ck<<<1, 128, 0, stream>>>(w2, beta2, mean2, var2, ckp);

    const int nblocks = (B_DIM * T_DIM) / TILE_T;
    resblock_shortcut_wmma<<<nblocks, NTHREADS, 0, stream>>>(x, wscb, ckp, (float*)d_out);
}
